// RNN_66718021976676
// MI455X (gfx1250) — matrix operand order repaired and checked by reading, not compiled, not run on hardware
//
#include <hip/hip_runtime.h>

// ---- CDNA5 WMMA vector types (probe-confirmed signatures) ----
typedef __attribute__((ext_vector_type(16))) __bf16 v16bf;
typedef __attribute__((ext_vector_type(8)))  float  v8f;

#define B_    512
#define S_    120
#define D_    216
#define H_    1024
#define G_    4096   // 4*H
#define DPAD  224    // D padded to multiple of 32 for K-loop
#define MAXLEN 24

// round-to-nearest-even float -> bf16 bits
static __device__ __forceinline__ unsigned short f2bf(float f) {
  unsigned u = __float_as_uint(f);
  u += 0x7FFFu + ((u >> 16) & 1u);
  return (unsigned short)(u >> 16);
}

// Load a 16-element bf16 fragment from two 16-byte chunks.
static __device__ __forceinline__ v16bf ldfrag(const unsigned short* p0,
                                               const unsigned short* p1) {
  union { v16bf v; uint4 q[2]; } f;
  f.q[0] = *(const uint4*)p0;
  f.q[1] = *(const uint4*)p1;
  return f.v;
}

static __device__ __forceinline__ float sigmoidf_(float x) {
  return 1.0f / (1.0f + __expf(-x));
}

// ---------------- prep kernels ----------------

__global__ __launch_bounds__(256) void prep_weights(
    const float* __restrict__ Wih, const float* __restrict__ Whh,
    const float* __restrict__ Wout, const float* __restrict__ bih,
    const float* __restrict__ bhh,
    unsigned short* __restrict__ Wih_b, unsigned short* __restrict__ Whh_b,
    unsigned short* __restrict__ Wout_b, float* __restrict__ bias) {
  const int n0 = G_ * DPAD;          // W_ih padded
  const int n1 = G_ * H_;            // W_hh
  const int n2 = DPAD * H_;          // W_out padded (rows)
  const int n3 = G_;                 // bias
  const int tot = n0 + n1 + n2 + n3;
  for (int i = blockIdx.x * blockDim.x + threadIdx.x; i < tot;
       i += gridDim.x * blockDim.x) {
    if (i < n0) {
      int r = i / DPAD, c = i - r * DPAD;
      Wih_b[i] = (c < D_) ? f2bf(Wih[r * D_ + c]) : (unsigned short)0;
    } else if (i < n0 + n1) {
      int j = i - n0;
      Whh_b[j] = f2bf(Whh[j]);
    } else if (i < n0 + n1 + n2) {
      int j = i - n0 - n1;
      int r = j / H_, c = j - r * H_;
      Wout_b[j] = (r < D_) ? f2bf(Wout[r * H_ + c]) : (unsigned short)0;
    } else {
      int j = i - n0 - n1 - n2;
      bias[j] = bih[j] + bhh[j];
    }
  }
}

__global__ __launch_bounds__(256) void prep_src(
    const float* __restrict__ src,          // (B, S, D)
    unsigned short* __restrict__ src_b,     // [S][B][DPAD]
    unsigned short* __restrict__ y_b,       // [B][DPAD]
    unsigned short* __restrict__ h_a,       // [B][H]
    unsigned short* __restrict__ h_bb,      // [B][H]
    float* __restrict__ c) {                // [B][H]
  const int n0 = S_ * B_ * DPAD;
  const int n1 = B_ * DPAD;
  const int n2 = B_ * H_;
  const int tot = n0 + n1 + 3 * n2;
  for (int i = blockIdx.x * blockDim.x + threadIdx.x; i < tot;
       i += gridDim.x * blockDim.x) {
    if (i < n0) {
      int s = i / (B_ * DPAD);
      int r = i - s * (B_ * DPAD);
      int b = r / DPAD, col = r - b * DPAD;
      src_b[i] = (col < D_) ? f2bf(src[(b * S_ + s) * D_ + col])
                            : (unsigned short)0;
    } else if (i < n0 + n1) {
      y_b[i - n0] = 0;
    } else if (i < n0 + n1 + n2) {
      h_a[i - n0 - n1] = 0;
    } else if (i < n0 + n1 + 2 * n2) {
      h_bb[i - n0 - n1 - n2] = 0;
    } else {
      c[i - n0 - n1 - 2 * n2] = 0.0f;
    }
  }
}

// ------------- fused LSTM step: dual GEMM + in-register cell update --------
// One wave computes 2 M-tiles x {i,f,g,o} gate columns for one 16-wide H patch:
// 8 f32 accumulators. A fragments reused 4x (across gates), B fragments 2x
// (across M-tiles). Cell nonlinearity applied in the epilogue — no gates
// buffer, no second kernel. h double-buffered (hin read / hout written).

__global__ __launch_bounds__(256) void lstm_step_fused(
    const unsigned short* __restrict__ x,     // [B_][DPAD] bf16
    const unsigned short* __restrict__ hin,   // [B_][H_]  bf16
    const unsigned short* __restrict__ Wih,   // [G_][DPAD] bf16
    const unsigned short* __restrict__ Whh,   // [G_][H_]  bf16
    const float* __restrict__ bias,           // [G_]
    float* __restrict__ c,                    // [B_][H_] f32 (in/out)
    unsigned short* __restrict__ hout) {      // [B_][H_] bf16
  const int lane = threadIdx.x & 31;
  const int wave = threadIdx.x >> 5;
  const int tile = blockIdx.x * 8 + wave;     // 1024 waves total
  const int nT   = H_ / 16;                   // 64 column tiles of H
  const int m0   = (tile / nT) * 32;          // 2 M-tiles per wave
  const int n0   = (tile % nT) * 16;          // column within H

  const int am  = lane & 15;                  // A row within tile
  const int akb = (lane >> 4) << 4;           // A K sub-block (0 or 16)
  const int bn  = lane & 15;                  // B column within tile
  const int bko = (lane >> 4) << 4;           // B K sub-block (0 or 16)

  v8f acc[2][4] = {};

  // ---- x @ Wih^T over padded K = 224 (4 gate columns) ----
  {
    const unsigned short* a0 = x + (m0 + am) * DPAD;
    const unsigned short* a1 = x + (m0 + 16 + am) * DPAD;
    const unsigned short* br0 = Wih + (n0 + 0 * H_ + bn) * DPAD;
    const unsigned short* br1 = Wih + (n0 + 1 * H_ + bn) * DPAD;
    const unsigned short* br2 = Wih + (n0 + 2 * H_ + bn) * DPAD;
    const unsigned short* br3 = Wih + (n0 + 3 * H_ + bn) * DPAD;
    for (int kk = 0; kk < DPAD; kk += 32) {
      v16bf A0 = ldfrag(a0 + kk + akb, a0 + kk + akb + 8);
      v16bf A1 = ldfrag(a1 + kk + akb, a1 + kk + akb + 8);
      v16bf Bf;
      Bf = ldfrag(br0 + kk + bko, br0 + kk + bko + 8);
      acc[0][0] = __builtin_amdgcn_wmma_f32_16x16x32_bf16(false, A0, false, Bf, (short)0, acc[0][0], false, false);
      acc[1][0] = __builtin_amdgcn_wmma_f32_16x16x32_bf16(false, A1, false, Bf, (short)0, acc[1][0], false, false);
      Bf = ldfrag(br1 + kk + bko, br1 + kk + bko + 8);
      acc[0][1] = __builtin_amdgcn_wmma_f32_16x16x32_bf16(false, A0, false, Bf, (short)0, acc[0][1], false, false);
      acc[1][1] = __builtin_amdgcn_wmma_f32_16x16x32_bf16(false, A1, false, Bf, (short)0, acc[1][1], false, false);
      Bf = ldfrag(br2 + kk + bko, br2 + kk + bko + 8);
      acc[0][2] = __builtin_amdgcn_wmma_f32_16x16x32_bf16(false, A0, false, Bf, (short)0, acc[0][2], false, false);
      acc[1][2] = __builtin_amdgcn_wmma_f32_16x16x32_bf16(false, A1, false, Bf, (short)0, acc[1][2], false, false);
      Bf = ldfrag(br3 + kk + bko, br3 + kk + bko + 8);
      acc[0][3] = __builtin_amdgcn_wmma_f32_16x16x32_bf16(false, A0, false, Bf, (short)0, acc[0][3], false, false);
      acc[1][3] = __builtin_amdgcn_wmma_f32_16x16x32_bf16(false, A1, false, Bf, (short)0, acc[1][3], false, false);
    }
  }
  // ---- h @ Whh^T over K = 1024 (4 gate columns) ----
  {
    const unsigned short* a0 = hin + (m0 + am) * H_;
    const unsigned short* a1 = hin + (m0 + 16 + am) * H_;
    const unsigned short* br0 = Whh + (n0 + 0 * H_ + bn) * H_;
    const unsigned short* br1 = Whh + (n0 + 1 * H_ + bn) * H_;
    const unsigned short* br2 = Whh + (n0 + 2 * H_ + bn) * H_;
    const unsigned short* br3 = Whh + (n0 + 3 * H_ + bn) * H_;
    for (int kk = 0; kk < H_; kk += 32) {
      __builtin_prefetch(br0 + kk + 256, 0, 1);   // global_prefetch_b8
      v16bf A0 = ldfrag(a0 + kk + akb, a0 + kk + akb + 8);
      v16bf A1 = ldfrag(a1 + kk + akb, a1 + kk + akb + 8);
      v16bf Bf;
      Bf = ldfrag(br0 + kk + bko, br0 + kk + bko + 8);
      acc[0][0] = __builtin_amdgcn_wmma_f32_16x16x32_bf16(false, A0, false, Bf, (short)0, acc[0][0], false, false);
      acc[1][0] = __builtin_amdgcn_wmma_f32_16x16x32_bf16(false, A1, false, Bf, (short)0, acc[1][0], false, false);
      Bf = ldfrag(br1 + kk + bko, br1 + kk + bko + 8);
      acc[0][1] = __builtin_amdgcn_wmma_f32_16x16x32_bf16(false, A0, false, Bf, (short)0, acc[0][1], false, false);
      acc[1][1] = __builtin_amdgcn_wmma_f32_16x16x32_bf16(false, A1, false, Bf, (short)0, acc[1][1], false, false);
      Bf = ldfrag(br2 + kk + bko, br2 + kk + bko + 8);
      acc[0][2] = __builtin_amdgcn_wmma_f32_16x16x32_bf16(false, A0, false, Bf, (short)0, acc[0][2], false, false);
      acc[1][2] = __builtin_amdgcn_wmma_f32_16x16x32_bf16(false, A1, false, Bf, (short)0, acc[1][2], false, false);
      Bf = ldfrag(br3 + kk + bko, br3 + kk + bko + 8);
      acc[0][3] = __builtin_amdgcn_wmma_f32_16x16x32_bf16(false, A0, false, Bf, (short)0, acc[0][3], false, false);
      acc[1][3] = __builtin_amdgcn_wmma_f32_16x16x32_bf16(false, A1, false, Bf, (short)0, acc[1][3], false, false);
    }
  }

  // ---- epilogue: in-register LSTM cell update for the 2x(16x16) patch ----
  const int cn  = lane & 15;
  const int cmb = (lane >> 4) << 3;            // M offset 0 or 8
  const float bI = bias[n0 + cn];
  const float bF = bias[n0 + H_ + cn];
  const float bG = bias[n0 + 2 * H_ + cn];
  const float bO = bias[n0 + 3 * H_ + cn];
#pragma unroll
  for (int mt = 0; mt < 2; ++mt) {
    const int rowbase = m0 + mt * 16 + cmb;
#pragma unroll
    for (int r = 0; r < 8; ++r) {
      const int idx = (rowbase + r) * H_ + n0 + cn;
      float si = sigmoidf_(acc[mt][0][r] + bI);
      float sf = sigmoidf_(acc[mt][1][r] + bF);
      float tg = tanhf(acc[mt][2][r] + bG);
      float so = sigmoidf_(acc[mt][3][r] + bO);
      float cnew = sf * c[idx] + si * tg;
      float hnew = so * tanhf(cnew);
      c[idx] = cnew;
      hout[idx] = f2bf(hnew);
    }
  }
}

// ---------------- output projection: y = h @ Wout^T + b_out ----------------
// M=512, N=224 (padded; guard col<216), K=1024.

__global__ __launch_bounds__(256) void proj_wmma(
    const unsigned short* __restrict__ h,     // [B_][H_] bf16
    const unsigned short* __restrict__ Wout,  // [DPAD][H_] bf16
    const float* __restrict__ b_out,          // [D_]
    float* __restrict__ out,                  // (B, MAXLEN, D)
    unsigned short* __restrict__ y_b,         // [B_][DPAD] bf16
    int step) {
  const int lane = threadIdx.x & 31;
  const int wave = threadIdx.x >> 5;
  const int tile = blockIdx.x * 8 + wave;     // 448 tiles
  const int tilesN = DPAD / 16;               // 14
  const int m0 = (tile / tilesN) << 4;
  const int n0 = (tile % tilesN) << 4;

  const int am  = lane & 15;
  const int akb = (lane >> 4) << 4;
  const int bn  = lane & 15;
  const int bko = (lane >> 4) << 4;

  v8f acc = {};
  const unsigned short* arow = h    + (m0 + am) * H_;
  const unsigned short* brow = Wout + (n0 + bn) * H_;
  for (int kk = 0; kk < H_; kk += 32) {
    v16bf a = ldfrag(arow + kk + akb, arow + kk + akb + 8);
    v16bf b = ldfrag(brow + kk + bko, brow + kk + bko + 8);
    acc = __builtin_amdgcn_wmma_f32_16x16x32_bf16(
        false, a, false, b, (short)0, acc, false, false);
  }

  const int cn  = lane & 15;
  const int cmb = (lane >> 4) << 3;
  const int col = n0 + cn;
  if (col < D_) {
    const float bv = b_out[col];
#pragma unroll
    for (int r = 0; r < 8; ++r) {
      int m = m0 + cmb + r;                   // batch index
      float v = acc[r] + bv;
      out[(m * MAXLEN + step) * D_ + col] = v;
      y_b[m * DPAD + col] = f2bf(v);
    }
  }
}

// ---------------- host launcher ----------------

extern "C" void kernel_launch(void* const* d_in, const int* in_sizes, int n_in,
                              void* d_out, int out_size, void* d_ws, size_t ws_size,
                              hipStream_t stream) {
  const float* src   = (const float*)d_in[0];
  const float* W_ih  = (const float*)d_in[1];
  const float* W_hh  = (const float*)d_in[2];
  const float* b_ih  = (const float*)d_in[3];
  const float* b_hh  = (const float*)d_in[4];
  const float* W_out = (const float*)d_in[5];
  const float* b_out = (const float*)d_in[6];
  float* out = (float*)d_out;

  unsigned char* ws = (unsigned char*)d_ws;
  // workspace layout (all 256B aligned), total ~40.7 MB
  unsigned short* Wih_b  = (unsigned short*)(ws + 0);             // 4096*224*2
  unsigned short* Whh_b  = (unsigned short*)(ws + 1835008);       // 4096*1024*2
  unsigned short* Wout_b = (unsigned short*)(ws + 10223616);      // 224*1024*2
  float*          bias   = (float*)         (ws + 10682368);      // 4096*4
  unsigned short* src_b  = (unsigned short*)(ws + 10698752);      // 120*512*224*2
  unsigned short* h_a    = (unsigned short*)(ws + 38223872);      // 512*1024*2
  unsigned short* h_bb   = (unsigned short*)(ws + 39272448);      // 512*1024*2
  float*          c_st   = (float*)         (ws + 40321024);      // 512*1024*4
  unsigned short* y_b    = (unsigned short*)(ws + 42418176);      // 512*224*2

  prep_weights<<<2048, 256, 0, stream>>>(W_ih, W_hh, W_out, b_ih, b_hh,
                                         Wih_b, Whh_b, Wout_b, bias);
  prep_src<<<4096, 256, 0, stream>>>(src, src_b, y_b, h_a, h_bb, c_st);

  unsigned short* hc = h_a;   // current h (read)
  unsigned short* hn = h_bb;  // next h (write)

  // encoder: teacher-forced over S steps
  for (int t = 0; t < S_; ++t) {
    const unsigned short* xt = src_b + (size_t)t * B_ * DPAD;
    lstm_step_fused<<<128, 256, 0, stream>>>(xt, hc, Wih_b, Whh_b, bias,
                                             c_st, hn);
    unsigned short* tmp = hc; hc = hn; hn = tmp;
  }

  // decoder: autoregressive, first input = src[:, S-1, :]
  const unsigned short* x0 = src_b + (size_t)(S_ - 1) * B_ * DPAD;
  for (int k = 0; k < MAXLEN; ++k) {
    const unsigned short* xk = (k == 0) ? x0 : (const unsigned short*)y_b;
    lstm_step_fused<<<128, 256, 0, stream>>>(xk, hc, Wih_b, Whh_b, bias,
                                             c_st, hn);
    unsigned short* tmp = hc; hc = hn; hn = tmp;
    proj_wmma<<<56, 256, 0, stream>>>(hc, Wout_b, b_out, out, y_b, k);
  }
}
